// VSSBlock_20126216749907
// MI455X (gfx1250) — matrix operand order repaired and checked by reading, not compiled, not run on hardware
//
#include <hip/hip_runtime.h>
#include <hip/hip_bf16.h>
#include <cstdint>

// ---------------------------------------------------------------------------
// VSSBlock for MI455X (gfx1250): bf16 WMMA GEMMs + chunked selective scan.
// Sizes (fixed by the reference): B=2, H=W=96, C=96, D_INNER=DM=192,
// D_STATE=16, DT_RANK=12, L=9216, tokens NT=18432.
// ---------------------------------------------------------------------------

static constexpr int NT   = 2 * 96 * 96;   // 18432 tokens
static constexpr int LSEQ = 96 * 96;       // 9216 per batch
static constexpr int DI   = 192;
static constexpr int XPW  = 44;            // dt(12) + B(16) + C(16)
static constexpr int NCHK = 144;           // chunks per batch
static constexpr int LCHK = 64;            // chunk length (144*64 = 9216)

typedef __attribute__((ext_vector_type(16))) __bf16 v16bf;
typedef __attribute__((ext_vector_type(8)))  float  v8f;

__device__ __forceinline__ float silu_f(float x) {
  return x * (1.f / (1.f + __expf(-x)));
}

// Load a 16-half fragment as two b128 LDS reads (both pointers 16B-aligned).
__device__ __forceinline__ v16bf ld_frag2(const __bf16* p0, const __bf16* p1) {
  union { uint4 u[2]; v16bf v; } t;
  t.u[0] = *reinterpret_cast<const uint4*>(p0);
  t.u[1] = *reinterpret_cast<const uint4*>(p1);
  return t.v;
}

// ---------------------------------------------------------------------------
// GEMM: C(MxN,f32) = A(MxK,bf16 row-major) * B(NxK,bf16 "transposed" weight).
// v_wmma_f32_16x16x32_bf16, fp32 accumulation. Block tile 128x32, 8 waves,
// each wave a 16x32 strip (two 16x16 accumulators). K % 32 == 0, M % 128 == 0;
// N guarded per-thread (handles N=44). EPI=1 adds a residual of C's shape.
//
// LDS tiles (padded stride 40 halves = 80B -> b128-aligned, conflict-free):
//   As[m][k]  : A fragment = two contiguous 8-half runs per lane (ISA layout
//               K = hf*8+0..7 and 16+hf*8+0..7) -> 2x ds_load_b128.
//   Bst[n][k] : B fragment = same K order as A (hf*8+0..7, 16+hf*8+0..7) -> 2x ds_load_b128.
// ---------------------------------------------------------------------------
template <int EPI>
__global__ __launch_bounds__(256) void k_gemm_bf16(
    const __bf16* __restrict__ Ag, const __bf16* __restrict__ Bt,
    float* __restrict__ Cg, const float* __restrict__ resid,
    int N, int K)
{
  __shared__ __bf16 As[128 * 40];
  __shared__ __bf16 Bst[32 * 40];

  const int tid  = threadIdx.x;
  const int wv   = tid >> 5;
  const int lane = tid & 31;
  const int r    = lane & 15;
  const int hf   = lane >> 4;
  const int bm   = blockIdx.x * 128;
  const int n0   = blockIdx.y * 32;

  v8f acc0 = {};
  v8f acc1 = {};

  // cooperative-load mappings
  const int am  = tid >> 1;          // A row within tile (0..127)
  const int aks = (tid & 1) * 16;    // A k-segment (16 halves = 32B)
  const int bn  = tid >> 3;          // B column within tile (0..31)
  const int bks = (tid & 7) * 4;     // B k-segment (4 halves = 8B)
  const bool bvalid = (n0 + bn) < N;

  const __bf16* afp0 = &As[(wv * 16 + r) * 40 + hf * 8];
  const __bf16* afp1 = afp0 + 16;
  const __bf16* b0p  = &Bst[r * 40 + hf * 8];
  const __bf16* b1p  = &Bst[(16 + r) * 40 + hf * 8];

  for (int k0 = 0; k0 < K; k0 += 32) {
    { // A tile: 128x32 bf16, one 32B run per thread
      const uint32_t* gp = reinterpret_cast<const uint32_t*>(
          Ag + (size_t)(bm + am) * K + k0 + aks);
      uint32_t* sp = reinterpret_cast<uint32_t*>(&As[am * 40 + aks]);
      #pragma unroll
      for (int i = 0; i < 8; ++i) sp[i] = gp[i];
    }
    { // B tile (transposed): one guarded 8B run per thread, zero-fill past N
      uint2 val = make_uint2(0u, 0u);
      if (bvalid)
        val = *reinterpret_cast<const uint2*>(Bt + (size_t)(n0 + bn) * K + k0 + bks);
      *reinterpret_cast<uint2*>(&Bst[bn * 40 + bks]) = val;
    }
    __syncthreads();

    if (k0 + 32 < K) {  // warm next A tile (global_prefetch_b8)
      __builtin_prefetch(Ag + (size_t)(bm + am) * K + k0 + 32 + aks, 0, 0);
    }

    v16bf af  = ld_frag2(afp0, afp1);
    v16bf bf0 = ld_frag2(b0p, b0p + 16);
    v16bf bf1 = ld_frag2(b1p, b1p + 16);

    acc0 = __builtin_amdgcn_wmma_f32_16x16x32_bf16(
        false, af, false, bf0, (short)0, acc0, false, false);
    acc1 = __builtin_amdgcn_wmma_f32_16x16x32_bf16(
        false, af, false, bf1, (short)0, acc1, false, false);
    __syncthreads();
  }

  // C layout: VGPR v holds M = hf*8 + v, N = lane%16
  const int col0 = n0 + r;
  const int col1 = n0 + 16 + r;
  #pragma unroll
  for (int v = 0; v < 8; ++v) {
    int row = bm + wv * 16 + hf * 8 + v;
    size_t o0 = (size_t)row * N + col0;
    size_t o1 = (size_t)row * N + col1;
    if (col0 < N) { float x = acc0[v]; if (EPI == 1) x += resid[o0]; Cg[o0] = x; }
    if (col1 < N) { float x = acc1[v]; if (EPI == 1) x += resid[o1]; Cg[o1] = x; }
  }
}

// ---------------------------------------------------------------------------
// Small utility kernels
// ---------------------------------------------------------------------------

// fp32 (KxN) -> bf16 transposed (NxK): dst[n*K + k] = src[k*N + n]
__global__ void k_cvt_bf16_t(const float* __restrict__ s, __bf16* __restrict__ d,
                             int K, int N) {
  int i = blockIdx.x * 256 + threadIdx.x;
  if (i < K * N) {
    int k = i % K;
    int n = i / K;
    d[i] = (__bf16)s[(size_t)k * N + n];
  }
}

__global__ void k_negexp(const float* __restrict__ s, float* __restrict__ d, int n) {
  int i = blockIdx.x * 256 + threadIdx.x;
  if (i < n) d[i] = -__expf(s[i]);
}

// LayerNorm over C=96: one wave per token (3 channels per lane, shfl reduce).
__global__ __launch_bounds__(256) void k_layernorm(
    const float* __restrict__ x, const float* __restrict__ w,
    const float* __restrict__ b, __bf16* __restrict__ out)
{
  const int t    = blockIdx.x * 8 + (threadIdx.x >> 5);
  const int lane = threadIdx.x & 31;
  const float* xr = x + (size_t)t * 96;
  float v0 = xr[lane], v1 = xr[lane + 32], v2 = xr[lane + 64];
  float s = v0 + v1 + v2;
  #pragma unroll
  for (int o = 16; o; o >>= 1) s += __shfl_xor(s, o);
  float mu = s * (1.f / 96.f);
  float d0 = v0 - mu, d1 = v1 - mu, d2 = v2 - mu;
  float q = d0 * d0 + d1 * d1 + d2 * d2;
  #pragma unroll
  for (int o = 16; o; o >>= 1) q += __shfl_xor(q, o);
  float rs = rsqrtf(q * (1.f / 96.f) + 1e-5f);
  __bf16* orow = out + (size_t)t * 96;
  orow[lane]      = (__bf16)(d0 * rs * w[lane]      + b[lane]);
  orow[lane + 32] = (__bf16)(d1 * rs * w[lane + 32] + b[lane + 32]);
  orow[lane + 64] = (__bf16)(d2 * rs * w[lane + 64] + b[lane + 64]);
}

// Depthwise 3x3 'same' conv over HxW + SiLU. Input = x_in (cols 0..191 of the
// 384-wide in_proj output); output bf16 (NT x 192).
__global__ __launch_bounds__(256) void k_dwconv2d_silu(
    const float* __restrict__ xio, const float* __restrict__ w2,
    __bf16* __restrict__ out)
{
  int idx = blockIdx.x * 256 + threadIdx.x;   // NT*192 threads exactly
  int d = idx % DI;
  int t = idx / DI;
  int wc = t % 96;
  int hr = (t / 96) % 96;
  int bb = t / (96 * 96);
  float acc = 0.f;
  #pragma unroll
  for (int kh = 0; kh < 3; ++kh) {
    int hh = hr + kh - 1;
    if (hh < 0 || hh >= 96) continue;
    #pragma unroll
    for (int kw = 0; kw < 3; ++kw) {
      int ww = wc + kw - 1;
      if (ww < 0 || ww >= 96) continue;
      int tt = (bb * 96 + hh) * 96 + ww;
      acc += xio[(size_t)tt * 384 + d] * w2[d * 9 + kh * 3 + kw];
    }
  }
  out[idx] = (__bf16)silu_f(acc);
}

// Causal depthwise conv1d (width 3) over L + bias + SiLU.
__global__ __launch_bounds__(256) void k_dwconv1d_silu(
    const float* __restrict__ mio, const float* __restrict__ w1,
    const float* __restrict__ b1, __bf16* __restrict__ out)
{
  int idx = blockIdx.x * 256 + threadIdx.x;   // NT*192
  int d = idx % DI;
  int t = idx / DI;
  int l = t % LSEQ;
  float acc = b1[d];
  #pragma unroll
  for (int j = 0; j < 3; ++j) {
    int ll = l + j - 2;
    if (ll >= 0) acc += mio[(size_t)(t + j - 2) * 384 + d] * w1[d * 3 + j];
  }
  out[idx] = (__bf16)silu_f(acc);
}

// delta = softplus(dt @ Wdt + bdt); dt = first 12 cols of x_dbl (ld 44).
__global__ __launch_bounds__(256) void k_dtproj_softplus(
    const float* __restrict__ xdbl, const float* __restrict__ Wdt,
    const float* __restrict__ bdt, float* __restrict__ delta)
{
  int idx = blockIdx.x * 256 + threadIdx.x;   // NT*192
  int d = idx % DI;
  int t = idx / DI;
  float acc = bdt[d];
  const float* xr = xdbl + (size_t)t * XPW;
  #pragma unroll
  for (int r = 0; r < 12; ++r) acc += xr[r] * Wdt[r * DI + d];
  delta[idx] = (acc > 20.f) ? acc : log1pf(__expf(acc));
}

// out_bf16 = y * silu(gate); gate lives in a 384-wide buffer at goff.
__global__ __launch_bounds__(256) void k_gate_silu_mul(
    const float* __restrict__ y, const float* __restrict__ g,
    int goff, __bf16* __restrict__ out)
{
  int idx = blockIdx.x * 256 + threadIdx.x;   // NT*192
  int d = idx % DI;
  int t = idx / DI;
  out[idx] = (__bf16)(y[idx] * silu_f(g[(size_t)t * 384 + goff + d]));
}

// ---------------------------------------------------------------------------
// Chunked selective scan. State h[d][n]: lanes 0-15 own d0 (n=lane), lanes
// 16-31 own d1 (n=lane-16). Transition product over a chunk = exp(A * sum d).
// ---------------------------------------------------------------------------
__global__ __launch_bounds__(256) void k_scan_pass1(
    const float* __restrict__ delta, const __bf16* __restrict__ u,
    const float* __restrict__ xdbl, const float* __restrict__ Aneg,
    float* __restrict__ hend, float* __restrict__ Ssum)
{
  const int lane = threadIdx.x & 31;
  const int wv   = threadIdx.x >> 5;
  const int hf   = lane >> 4;
  const int n    = lane & 15;
  const int d    = blockIdx.y * 16 + wv * 2 + hf;
  const int c    = blockIdx.x;
  const int bb   = blockIdx.z;
  const float Adn = Aneg[d * 16 + n];
  float h = 0.f, s = 0.f;
  const int t0 = bb * LSEQ + c * LCHK;
  for (int l = 0; l < LCHK; ++l) {
    const size_t t = (size_t)(t0 + l);
    float dl = delta[t * DI + d];
    float ul = (float)u[t * DI + d];
    float bn = xdbl[t * XPW + 12 + n];
    h = __expf(Adn * dl) * h + (dl * ul) * bn;
    s += dl;
  }
  size_t base = ((size_t)bb * NCHK + c) * DI + d;
  hend[base * 16 + n] = h;
  if (n == 0) Ssum[base] = s;
}

__global__ __launch_bounds__(256) void k_scan_pass2(
    const float* __restrict__ Ssum, const float* __restrict__ hend,
    const float* __restrict__ Aneg, float* __restrict__ hin)
{
  int tid = blockIdx.x * 256 + threadIdx.x;   // 2*192*16 = 6144
  int n = tid & 15;
  int d = (tid >> 4) % DI;
  int bb = tid / (DI * 16);
  float Adn = Aneg[d * 16 + n];
  float h = 0.f;
  for (int c = 0; c < NCHK; ++c) {
    size_t base = ((size_t)bb * NCHK + c) * DI + d;
    hin[base * 16 + n] = h;
    h = __expf(Adn * Ssum[base]) * h + hend[base * 16 + n];
  }
}

__global__ __launch_bounds__(256) void k_scan_pass3(
    const float* __restrict__ delta, const __bf16* __restrict__ u,
    const float* __restrict__ xdbl, const float* __restrict__ Aneg,
    const float* __restrict__ hin, const float* __restrict__ Dv,
    float* __restrict__ yscan)
{
  const int lane = threadIdx.x & 31;
  const int wv   = threadIdx.x >> 5;
  const int hf   = lane >> 4;
  const int n    = lane & 15;
  const int d    = blockIdx.y * 16 + wv * 2 + hf;
  const int c    = blockIdx.x;
  const int bb   = blockIdx.z;
  size_t base = ((size_t)bb * NCHK + c) * DI + d;
  float h = hin[base * 16 + n];
  const float Adn = Aneg[d * 16 + n];
  const float dd  = Dv[d];
  const int t0 = bb * LSEQ + c * LCHK;
  for (int l = 0; l < LCHK; ++l) {
    const size_t t = (size_t)(t0 + l);
    float dl = delta[t * DI + d];
    float ul = (float)u[t * DI + d];
    float bn = xdbl[t * XPW + 12 + n];
    float cn = xdbl[t * XPW + 28 + n];
    h = __expf(Adn * dl) * h + (dl * ul) * bn;
    float p = h * cn;                      // reduce over 16 states (half-wave)
    p += __shfl_xor(p, 8);
    p += __shfl_xor(p, 4);
    p += __shfl_xor(p, 2);
    p += __shfl_xor(p, 1);
    if (n == 0) yscan[t * DI + d] = p + ul * dd;
  }
}

// ---------------------------------------------------------------------------
// Host orchestration
// ---------------------------------------------------------------------------
extern "C" void kernel_launch(void* const* d_in, const int* in_sizes, int n_in,
                              void* d_out, int out_size, void* d_ws, size_t ws_size,
                              hipStream_t stream)
{
  (void)in_sizes; (void)n_in; (void)out_size; (void)ws_size;
  const float* x           = (const float*)d_in[0];
  const float* norm_w      = (const float*)d_in[1];
  const float* norm_b      = (const float*)d_in[2];
  const float* in_proj_w   = (const float*)d_in[3];
  const float* conv2d_w    = (const float*)d_in[4];
  const float* m_in_proj_w = (const float*)d_in[5];
  const float* m_conv1d_w  = (const float*)d_in[6];
  const float* m_conv1d_b  = (const float*)d_in[7];
  const float* m_x_proj_w  = (const float*)d_in[8];
  const float* m_dt_proj_w = (const float*)d_in[9];
  const float* m_dt_proj_b = (const float*)d_in[10];
  const float* m_A_log     = (const float*)d_in[11];
  const float* m_D         = (const float*)d_in[12];
  const float* m_out_proj_w= (const float*)d_in[13];
  const float* out_proj_w  = (const float*)d_in[14];
  float* out = (float*)d_out;

  size_t off = 0;
  auto alloc = [&](size_t bytes) -> void* {
    void* p = (char*)d_ws + off;
    off = (off + bytes + 255) & ~(size_t)255;
    return p;
  };

  // weights stored bf16 *transposed* (N x K) for the GEMM's B-tile path
  __bf16* wbf_in   = (__bf16*)alloc((size_t)96 * 384 * 2);
  __bf16* wbf_min  = (__bf16*)alloc((size_t)192 * 384 * 2);
  __bf16* wbf_x    = (__bf16*)alloc((size_t)192 * XPW * 2);
  __bf16* wbf_outm = (__bf16*)alloc((size_t)192 * 192 * 2);
  __bf16* wbf_proj = (__bf16*)alloc((size_t)192 * 96 * 2);
  float*  Aneg     = (float*)alloc((size_t)192 * 16 * 4);
  __bf16* xn_bf    = (__bf16*)alloc((size_t)NT * 96 * 2);
  float*  xio      = (float*)alloc((size_t)NT * 384 * 4);   // [x_in | x_gate]
  __bf16* xc_bf    = (__bf16*)alloc((size_t)NT * DI * 2);
  float*  mio      = (float*)alloc((size_t)NT * 384 * 4);   // [xm | z]
  __bf16* u_bf     = (__bf16*)alloc((size_t)NT * DI * 2);
  float*  xdbl     = (float*)alloc((size_t)NT * XPW * 4);
  float*  delta    = (float*)alloc((size_t)NT * DI * 4);
  float*  Ssum     = (float*)alloc((size_t)2 * NCHK * DI * 4);
  float*  hend     = (float*)alloc((size_t)2 * NCHK * DI * 16 * 4);
  float*  hin      = (float*)alloc((size_t)2 * NCHK * DI * 16 * 4);
  float*  yscan    = (float*)alloc((size_t)NT * DI * 4);
  __bf16* ygated   = (__bf16*)alloc((size_t)NT * DI * 2);
  float*  ymamba   = (float*)alloc((size_t)NT * DI * 4);
  __bf16* yg2_bf   = (__bf16*)alloc((size_t)NT * DI * 2);

  const int ntd = NT * DI;                 // 3538944
  const dim3 blk(256);

  // --- weight prep: bf16 transposed copies + A = -exp(A_log) ---
  k_cvt_bf16_t<<<(96 * 384 + 255) / 256, blk, 0, stream>>>(in_proj_w, wbf_in, 96, 384);
  k_cvt_bf16_t<<<(192 * 384 + 255) / 256, blk, 0, stream>>>(m_in_proj_w, wbf_min, 192, 384);
  k_cvt_bf16_t<<<(192 * XPW + 255) / 256, blk, 0, stream>>>(m_x_proj_w, wbf_x, 192, XPW);
  k_cvt_bf16_t<<<(192 * 192 + 255) / 256, blk, 0, stream>>>(m_out_proj_w, wbf_outm, 192, 192);
  k_cvt_bf16_t<<<(192 * 96 + 255) / 256, blk, 0, stream>>>(out_proj_w, wbf_proj, 192, 96);
  k_negexp<<<(192 * 16 + 255) / 256, blk, 0, stream>>>(m_A_log, Aneg, 192 * 16);

  // --- LN -> in_proj GEMM (18432x96 @ 96x384) ---
  k_layernorm<<<NT / 8, blk, 0, stream>>>(x, norm_w, norm_b, xn_bf);
  k_gemm_bf16<0><<<dim3(NT / 128, 12), blk, 0, stream>>>(xn_bf, wbf_in, xio, nullptr, 384, 96);

  // --- depthwise 3x3 conv + SiLU -> seq ---
  k_dwconv2d_silu<<<ntd / 256, blk, 0, stream>>>(xio, conv2d_w, xc_bf);

  // --- mamba in_proj (18432x192 @ 192x384) ---
  k_gemm_bf16<0><<<dim3(NT / 128, 12), blk, 0, stream>>>(xc_bf, wbf_min, mio, nullptr, 384, 192);

  // --- causal conv1d + SiLU -> u ---
  k_dwconv1d_silu<<<ntd / 256, blk, 0, stream>>>(mio, m_conv1d_w, m_conv1d_b, u_bf);

  // --- x_proj (18432x192 @ 192x44) ---
  k_gemm_bf16<0><<<dim3(NT / 128, 2), blk, 0, stream>>>(u_bf, wbf_x, xdbl, nullptr, XPW, 192);

  // --- dt_proj + softplus ---
  k_dtproj_softplus<<<ntd / 256, blk, 0, stream>>>(xdbl, m_dt_proj_w, m_dt_proj_b, delta);

  // --- chunked selective scan ---
  k_scan_pass1<<<dim3(NCHK, DI / 16, 2), blk, 0, stream>>>(delta, u_bf, xdbl, Aneg, hend, Ssum);
  k_scan_pass2<<<(2 * DI * 16) / 256, blk, 0, stream>>>(Ssum, hend, Aneg, hin);
  k_scan_pass3<<<dim3(NCHK, DI / 16, 2), blk, 0, stream>>>(delta, u_bf, xdbl, Aneg, hin, m_D, yscan);

  // --- y * silu(z) -> mamba out_proj (192x192) ---
  k_gate_silu_mul<<<ntd / 256, blk, 0, stream>>>(yscan, mio, 192, ygated);
  k_gemm_bf16<0><<<dim3(NT / 128, 6), blk, 0, stream>>>(ygated, wbf_outm, ymamba, nullptr, 192, 192);

  // --- * silu(x_gate) -> final out_proj (192x96) + residual x ---
  k_gate_silu_mul<<<ntd / 256, blk, 0, stream>>>(ymamba, xio, 192, yg2_bf);
  k_gemm_bf16<1><<<dim3(NT / 128, 3), blk, 0, stream>>>(yg2_bf, wbf_proj, out, x, 96, 192);
}
